// MoELayer_15745350107277
// MI455X (gfx1250) — compile-verified, matrix operand order checked by reading, not run on hardware
//
#include <hip/hip_runtime.h>
#include <hip/hip_bf16.h>
#include <stdint.h>

typedef __attribute__((ext_vector_type(16))) _Float16 v16h;
typedef __attribute__((ext_vector_type(8)))  float    v8f;

#define D_MODEL   1024
#define D_FF      2048
#define N_EXPERTS 8
#define BT        4096

// ---------------- workspace layout (bytes) ----------------
static constexpr size_t SZ_XH   = (size_t)BT * D_MODEL * 2;                 // x in f16
static constexpr size_t SZ_W1T  = (size_t)N_EXPERTS * D_FF * D_MODEL * 2;   // [E][F][D]
static constexpr size_t SZ_W3T  = SZ_W1T;
static constexpr size_t SZ_W2T  = (size_t)N_EXPERTS * D_MODEL * D_FF * 2;   // [E][D][F]
static constexpr size_t SZ_WS1T = (size_t)D_FF * D_MODEL * 2;               // [F][D]
static constexpr size_t SZ_WS3T = SZ_WS1T;
static constexpr size_t SZ_WS2T = (size_t)D_MODEL * D_FF * 2;               // [D][F]
static constexpr size_t SZ_H    = (size_t)2 * BT * D_FF * 2;                // 8192 slots
static constexpr size_t SZ_HS   = (size_t)BT * D_FF * 2;                    // shared hidden
static constexpr size_t SZ_Y    = (size_t)2 * BT * D_MODEL * 2;             // per-slot y (f16)
static constexpr size_t SZ_TOK  = (size_t)N_EXPERTS * BT * 4;               // token lists
static constexpr size_t SZ_CNT  = 256;
static constexpr size_t SZ_OFFS = 256;
static constexpr size_t SZ_TEXP = (size_t)BT * 4 * 4;                       // (e0,p0,e1,p1)
static constexpr size_t SZ_TW   = (size_t)BT * 2 * 4;                       // (w0,w1)

static constexpr size_t OFF_XH   = 0;
static constexpr size_t OFF_W1T  = OFF_XH   + SZ_XH;
static constexpr size_t OFF_W3T  = OFF_W1T  + SZ_W1T;
static constexpr size_t OFF_W2T  = OFF_W3T  + SZ_W3T;
static constexpr size_t OFF_WS1T = OFF_W2T  + SZ_W2T;
static constexpr size_t OFF_WS3T = OFF_WS1T + SZ_WS1T;
static constexpr size_t OFF_WS2T = OFF_WS3T + SZ_WS3T;
static constexpr size_t OFF_H    = OFF_WS2T + SZ_WS2T;
static constexpr size_t OFF_HS   = OFF_H    + SZ_H;
static constexpr size_t OFF_Y    = OFF_HS   + SZ_HS;
static constexpr size_t OFF_TOK  = OFF_Y    + SZ_Y;
static constexpr size_t OFF_CNT  = OFF_TOK  + SZ_TOK;
static constexpr size_t OFF_OFFS = OFF_CNT  + SZ_CNT;
static constexpr size_t OFF_TEXP = OFF_OFFS + SZ_OFFS;
static constexpr size_t OFF_TW   = OFF_TEXP + SZ_TEXP;

// output segment offsets (elements) in d_out: out, full_probs, logits, topk_idx
static constexpr size_t OUT_Y     = 0;
static constexpr size_t OUT_PROBS = (size_t)BT * D_MODEL;
static constexpr size_t OUT_LOGIT = OUT_PROBS + (size_t)BT * N_EXPERTS;
static constexpr size_t OUT_IDX   = OUT_LOGIT + (size_t)BT * N_EXPERTS;

// ---------------- fragment helpers ----------------
union Frag { v16h h; uint4 q[2]; };

// Load a 16-f16 WMMA A/B fragment for this lane: two contiguous 16-byte K
// chunks per the CDNA5 16-bit 16x32 layout (lanes<16: K 0-7,16-23;
// lanes>=16: K 8-15,24-31).
__device__ __forceinline__ v16h load_frag(const _Float16* __restrict__ rowk, int laneHi) {
    Frag f;
    const _Float16* b0 = rowk + (laneHi ? 8 : 0);
    f.q[0] = *(const uint4*)(b0);
    f.q[1] = *(const uint4*)(b0 + 16);
    return f.h;
}

__device__ __forceinline__ v8f wmma_f16(v16h a, v16h b, v8f c) {
    return __builtin_amdgcn_wmma_f32_16x16x32_f16(false, a, false, b, (short)0, c,
                                                  false, false);
}

// ---------------- small utility kernels ----------------
__global__ void zero_ints(int* p, int n) {
    int i = blockIdx.x * blockDim.x + threadIdx.x;
    if (i < n) p[i] = 0;
}

__global__ void cvt_f16(const float* __restrict__ src, _Float16* __restrict__ dst, int n) {
    int i = (blockIdx.x * blockDim.x + threadIdx.x) * 4;
    if (i < n) {
        float4 v = *(const float4*)(src + i);
        dst[i + 0] = (_Float16)v.x;
        dst[i + 1] = (_Float16)v.y;
        dst[i + 2] = (_Float16)v.z;
        dst[i + 3] = (_Float16)v.w;
    }
}

// fp32 [batch][R][C] -> f16 [batch][C][R]; R,C multiples of 32
__global__ void transpose_cvt(const float* __restrict__ src, _Float16* __restrict__ dst,
                              int R, int C) {
    __shared__ _Float16 tile[32][33];
    size_t bo = (size_t)blockIdx.z * R * C;
    const float* s = src + bo;
    _Float16*    d = dst + bo;
    int c0 = blockIdx.x * 32, r0 = blockIdx.y * 32;
    int tx = threadIdx.x, ty = threadIdx.y;
    #pragma unroll
    for (int i = 0; i < 32; i += 8)
        tile[ty + i][tx] = (_Float16)s[(size_t)(r0 + ty + i) * C + c0 + tx];
    __syncthreads();
    #pragma unroll
    for (int i = 0; i < 32; i += 8)
        d[(size_t)(c0 + ty + i) * R + r0 + tx] = tile[tx][ty + i];
}

// ---------------- router: logits, softmax, top-2, dispatch lists ----------------
__global__ void router_kernel(const float* __restrict__ x, const float* __restrict__ Wr,
                              float* __restrict__ outLogits, float* __restrict__ outProbs,
                              int* __restrict__ outIdx,
                              int* __restrict__ cnt, int* __restrict__ tokens,
                              int* __restrict__ tokexp, float* __restrict__ tokw) {
    int wave = threadIdx.x >> 5;
    int lane = threadIdx.x & 31;
    int t = blockIdx.x * 4 + wave;
    if (t >= BT) return;
    float acc[N_EXPERTS];
    #pragma unroll
    for (int e = 0; e < N_EXPERTS; e++) acc[e] = 0.f;
    const float* xr = x + (size_t)t * D_MODEL;
    for (int j = 0; j < D_MODEL / 32; j++) {
        float xv = xr[j * 32 + lane];
        const float* wr = Wr + (size_t)(j * 32 + lane) * N_EXPERTS;
        #pragma unroll
        for (int e = 0; e < N_EXPERTS; e++) acc[e] += xv * wr[e];
    }
    #pragma unroll
    for (int e = 0; e < N_EXPERTS; e++)
        for (int off = 16; off > 0; off >>= 1)
            acc[e] += __shfl_xor(acc[e], off, 32);
    if (lane == 0) {
        float mx = acc[0];
        #pragma unroll
        for (int e = 1; e < N_EXPERTS; e++) mx = fmaxf(mx, acc[e]);
        float p[N_EXPERTS], s = 0.f;
        #pragma unroll
        for (int e = 0; e < N_EXPERTS; e++) { p[e] = __expf(acc[e] - mx); s += p[e]; }
        float inv = 1.f / s;
        #pragma unroll
        for (int e = 0; e < N_EXPERTS; e++) {
            p[e] *= inv;
            outProbs[(size_t)t * N_EXPERTS + e]  = p[e];
            outLogits[(size_t)t * N_EXPERTS + e] = acc[e];
        }
        int i0 = 0;
        #pragma unroll
        for (int e = 1; e < N_EXPERTS; e++) if (p[e] > p[i0]) i0 = e;
        int i1 = (i0 == 0) ? 1 : 0;
        #pragma unroll
        for (int e = 0; e < N_EXPERTS; e++) if (e != i0 && p[e] > p[i1]) i1 = e;
        float w0 = p[i0], w1 = p[i1], wsum = w0 + w1;
        w0 /= wsum; w1 /= wsum;
        outIdx[t * 2 + 0] = i0;
        outIdx[t * 2 + 1] = i1;
        int p0 = atomicAdd(&cnt[i0], 1); tokens[i0 * BT + p0] = t;
        int p1 = atomicAdd(&cnt[i1], 1); tokens[i1 * BT + p1] = t;
        tokexp[t * 4 + 0] = i0; tokexp[t * 4 + 1] = p0;
        tokexp[t * 4 + 2] = i1; tokexp[t * 4 + 3] = p1;
        tokw[t * 2 + 0] = w0;   tokw[t * 2 + 1] = w1;
    }
}

__global__ void scan_offsets(const int* __restrict__ cnt, int* __restrict__ offs) {
    if (threadIdx.x == 0 && blockIdx.x == 0) {
        int s = 0;
        for (int e = 0; e < N_EXPERTS; e++) { offs[e] = s; s += cnt[e]; }
        offs[N_EXPERTS] = s;
    }
}

// ---------------- FFN1: h = silu(x@W1) * (x@W3), gathered rows ----------------
// Block tile: 128(M) x 32(N); each wave owns 32 rows (2 A-frags) so every B
// fragment feeds 2 WMMAs (12 b128 loads per 8 WMMAs instead of 18).
__global__ void ffn1_kernel(const _Float16* __restrict__ xh,
                            const _Float16* __restrict__ W1t,  // [E?][F][D]
                            const _Float16* __restrict__ W3t,
                            _Float16* __restrict__ hout,
                            const int* __restrict__ tokens,
                            const int* __restrict__ cnt,
                            const int* __restrict__ offs,
                            int sharedMode) {
    int e = blockIdx.z;
    int count = sharedMode ? BT : cnt[e];
    int mBlock = blockIdx.y * 128;
    if (mBlock >= count) return;
    int wave = threadIdx.x >> 5, lane = threadIdx.x & 31;
    int laneHi = lane >> 4, laneLo = lane & 15;
    int mWave = mBlock + wave * 32;
    int mm0 = mWave + laneLo;      if (mm0 >= count) mm0 = count - 1;
    int mm1 = mWave + 16 + laneLo; if (mm1 >= count) mm1 = count - 1;
    int srcRow0 = sharedMode ? mm0 : tokens[e * BT + mm0];
    int srcRow1 = sharedMode ? mm1 : tokens[e * BT + mm1];
    const _Float16* aRow0 = xh + (size_t)srcRow0 * D_MODEL;
    const _Float16* aRow1 = xh + (size_t)srcRow1 * D_MODEL;
    size_t we = sharedMode ? 0 : (size_t)e * D_FF * D_MODEL;
    const _Float16* w1e = W1t + we;
    const _Float16* w3e = W3t + we;
    int n0 = blockIdx.x * 32;
    v8f z = {};
    v8f acc1[2][2] = { { z, z }, { z, z } };
    v8f acc3[2][2] = { { z, z }, { z, z } };
    for (int kk = 0; kk < D_MODEL; kk += 32) {
        v16h a0 = load_frag(aRow0 + kk, laneHi);
        v16h a1 = load_frag(aRow1 + kk, laneHi);
        #pragma unroll
        for (int nt = 0; nt < 2; nt++) {
            int n = n0 + nt * 16 + laneLo;
            v16h b1 = load_frag(w1e + (size_t)n * D_MODEL + kk, laneHi);
            acc1[0][nt] = wmma_f16(a0, b1, acc1[0][nt]);
            acc1[1][nt] = wmma_f16(a1, b1, acc1[1][nt]);
            v16h b3 = load_frag(w3e + (size_t)n * D_MODEL + kk, laneHi);
            acc3[0][nt] = wmma_f16(a0, b3, acc3[0][nt]);
            acc3[1][nt] = wmma_f16(a1, b3, acc3[1][nt]);
        }
    }
    int rowBase = sharedMode ? 0 : offs[e];
    #pragma unroll
    for (int mi = 0; mi < 2; mi++) {
        #pragma unroll
        for (int nt = 0; nt < 2; nt++) {
            int col = n0 + nt * 16 + laneLo;
            #pragma unroll
            for (int r = 0; r < 8; r++) {
                int m = mWave + mi * 16 + r + laneHi * 8;
                if (m < count) {
                    float g = acc1[mi][nt][r];
                    float sg = g / (1.f + __expf(-g));
                    hout[(size_t)(rowBase + m) * D_FF + col] =
                        (_Float16)(sg * acc3[mi][nt][r]);
                }
            }
        }
    }
}

// ---------------- FFN2: y = h @ W2 (experts -> y f16; shared -> out f32) -----
__global__ void ffn2_kernel(const _Float16* __restrict__ hbase,
                            const _Float16* __restrict__ W2t,  // [E?][D][F]
                            _Float16* __restrict__ yout,
                            float* __restrict__ outF,
                            const int* __restrict__ cnt,
                            const int* __restrict__ offs,
                            int sharedMode) {
    int e = blockIdx.z;
    int count = sharedMode ? BT : cnt[e];
    int mBlock = blockIdx.y * 128;
    if (mBlock >= count) return;
    int wave = threadIdx.x >> 5, lane = threadIdx.x & 31;
    int laneHi = lane >> 4, laneLo = lane & 15;
    int mWave = mBlock + wave * 32;
    int mm0 = mWave + laneLo;      if (mm0 >= count) mm0 = count - 1;
    int mm1 = mWave + 16 + laneLo; if (mm1 >= count) mm1 = count - 1;
    int rowBase = sharedMode ? 0 : offs[e];
    const _Float16* aRow0 = hbase + (size_t)(rowBase + mm0) * D_FF;
    const _Float16* aRow1 = hbase + (size_t)(rowBase + mm1) * D_FF;
    const _Float16* w2e = W2t + (sharedMode ? 0 : (size_t)e * D_MODEL * D_FF);
    int n0 = blockIdx.x * 32;
    v8f z = {};
    v8f acc[2][2] = { { z, z }, { z, z } };
    for (int kk = 0; kk < D_FF; kk += 32) {
        v16h a0 = load_frag(aRow0 + kk, laneHi);
        v16h a1 = load_frag(aRow1 + kk, laneHi);
        #pragma unroll
        for (int nt = 0; nt < 2; nt++) {
            int n = n0 + nt * 16 + laneLo;
            v16h b = load_frag(w2e + (size_t)n * D_FF + kk, laneHi);
            acc[0][nt] = wmma_f16(a0, b, acc[0][nt]);
            acc[1][nt] = wmma_f16(a1, b, acc[1][nt]);
        }
    }
    #pragma unroll
    for (int mi = 0; mi < 2; mi++) {
        #pragma unroll
        for (int nt = 0; nt < 2; nt++) {
            int col = n0 + nt * 16 + laneLo;
            #pragma unroll
            for (int r = 0; r < 8; r++) {
                int m = mWave + mi * 16 + r + laneHi * 8;
                if (m < count) {
                    if (sharedMode)
                        outF[(size_t)m * D_MODEL + col] = acc[mi][nt][r];
                    else
                        yout[(size_t)(rowBase + m) * D_MODEL + col] =
                            (_Float16)acc[mi][nt][r];
                }
            }
        }
    }
}

// ---------------- combine: out[t] += w0*y[slot0] + w1*y[slot1] ----------------
__global__ void combine_kernel(float* __restrict__ out, const _Float16* __restrict__ y,
                               const int* __restrict__ tokexp, const float* __restrict__ tokw,
                               const int* __restrict__ offs) {
    int id = blockIdx.x * blockDim.x + threadIdx.x;  // one thread per 4 floats
    int t = id >> 8;
    int d = (id & 255) * 4;
    if (t >= BT) return;
    int e0 = tokexp[t * 4 + 0], p0 = tokexp[t * 4 + 1];
    int e1 = tokexp[t * 4 + 2], p1 = tokexp[t * 4 + 3];
    float w0 = tokw[t * 2 + 0], w1 = tokw[t * 2 + 1];
    const _Float16* y0 = y + (size_t)(offs[e0] + p0) * D_MODEL + d;
    const _Float16* y1 = y + (size_t)(offs[e1] + p1) * D_MODEL + d;
    float* op = out + (size_t)t * D_MODEL + d;
    float4 o = *(float4*)op;
    o.x += w0 * (float)y0[0] + w1 * (float)y1[0];
    o.y += w0 * (float)y0[1] + w1 * (float)y1[1];
    o.z += w0 * (float)y0[2] + w1 * (float)y1[2];
    o.w += w0 * (float)y0[3] + w1 * (float)y1[3];
    *(float4*)op = o;
}

// ---------------- host launch ----------------
extern "C" void kernel_launch(void* const* d_in, const int* in_sizes, int n_in,
                              void* d_out, int out_size, void* d_ws, size_t ws_size,
                              hipStream_t stream) {
    const float* x   = (const float*)d_in[0];
    const float* Wr  = (const float*)d_in[1];
    const float* W1  = (const float*)d_in[2];
    const float* W3  = (const float*)d_in[3];
    const float* W2  = (const float*)d_in[4];
    const float* Ws1 = (const float*)d_in[5];
    const float* Ws3 = (const float*)d_in[6];
    const float* Ws2 = (const float*)d_in[7];

    char* ws = (char*)d_ws;
    _Float16* xh   = (_Float16*)(ws + OFF_XH);
    _Float16* W1t  = (_Float16*)(ws + OFF_W1T);
    _Float16* W3t  = (_Float16*)(ws + OFF_W3T);
    _Float16* W2t  = (_Float16*)(ws + OFF_W2T);
    _Float16* Ws1t = (_Float16*)(ws + OFF_WS1T);
    _Float16* Ws3t = (_Float16*)(ws + OFF_WS3T);
    _Float16* Ws2t = (_Float16*)(ws + OFF_WS2T);
    _Float16* h    = (_Float16*)(ws + OFF_H);
    _Float16* hs   = (_Float16*)(ws + OFF_HS);
    _Float16* y    = (_Float16*)(ws + OFF_Y);
    int*   tokens  = (int*)(ws + OFF_TOK);
    int*   cnt     = (int*)(ws + OFF_CNT);
    int*   offs    = (int*)(ws + OFF_OFFS);
    int*   tokexp  = (int*)(ws + OFF_TEXP);
    float* tokw    = (float*)(ws + OFF_TW);

    float* outY     = (float*)d_out + OUT_Y;
    float* outProbs = (float*)d_out + OUT_PROBS;
    float* outLogit = (float*)d_out + OUT_LOGIT;
    int*   outIdx   = (int*)((float*)d_out + OUT_IDX);

    // 1. zero dispatch counts
    zero_ints<<<1, 32, 0, stream>>>(cnt, N_EXPERTS);

    // 2. convert x to f16
    {
        int n = BT * D_MODEL;
        cvt_f16<<<(n / 4 + 255) / 256, 256, 0, stream>>>(x, xh, n);
    }

    // 3. transpose+convert weights: expert GEMMs want [N][K] f16
    dim3 tb(32, 8);
    transpose_cvt<<<dim3(D_FF / 32, D_MODEL / 32, N_EXPERTS), tb, 0, stream>>>(W1, W1t, D_MODEL, D_FF);
    transpose_cvt<<<dim3(D_FF / 32, D_MODEL / 32, N_EXPERTS), tb, 0, stream>>>(W3, W3t, D_MODEL, D_FF);
    transpose_cvt<<<dim3(D_MODEL / 32, D_FF / 32, N_EXPERTS), tb, 0, stream>>>(W2, W2t, D_FF, D_MODEL);
    transpose_cvt<<<dim3(D_FF / 32, D_MODEL / 32, 1), tb, 0, stream>>>(Ws1, Ws1t, D_MODEL, D_FF);
    transpose_cvt<<<dim3(D_FF / 32, D_MODEL / 32, 1), tb, 0, stream>>>(Ws3, Ws3t, D_MODEL, D_FF);
    transpose_cvt<<<dim3(D_MODEL / 32, D_FF / 32, 1), tb, 0, stream>>>(Ws2, Ws2t, D_FF, D_MODEL);

    // 4. router
    router_kernel<<<BT / 4, 128, 0, stream>>>(x, Wr, outLogit, outProbs, outIdx,
                                              cnt, tokens, tokexp, tokw);

    // 5. exclusive scan of counts
    scan_offsets<<<1, 32, 0, stream>>>(cnt, offs);

    // 6. FFN1 (experts + shared)
    ffn1_kernel<<<dim3(D_FF / 32, BT / 128, N_EXPERTS), 128, 0, stream>>>(
        xh, W1t, W3t, h, tokens, cnt, offs, 0);
    ffn1_kernel<<<dim3(D_FF / 32, BT / 128, 1), 128, 0, stream>>>(
        xh, Ws1t, Ws3t, hs, tokens, cnt, offs, 1);

    // 7. FFN2: shared writes out directly; experts write per-slot y
    ffn2_kernel<<<dim3(D_MODEL / 32, BT / 128, 1), 128, 0, stream>>>(
        hs, Ws2t, y, outY, cnt, offs, 1);
    ffn2_kernel<<<dim3(D_MODEL / 32, BT / 128, N_EXPERTS), 128, 0, stream>>>(
        h, W2t, y, outY, cnt, offs, 0);

    // 8. combine routed experts into out
    combine_kernel<<<(BT * 256 + 255) / 256, 256, 0, stream>>>(outY, y, tokexp, tokw, offs);

    (void)in_sizes; (void)n_in; (void)out_size; (void)ws_size;
}
